// MultiHeaded2MultiheadAttentionGRU_1365799600445
// MI455X (gfx1250) — compile-verified, matrix operand order checked by reading, not run on hardware
//
#include <hip/hip_runtime.h>

// ---------------------------------------------------------------------------
// Problem constants (match reference)
// ---------------------------------------------------------------------------
#define B_ 32
#define S_ 512
#define H_ 512
#define H3_ 1536
#define NH_ 8
#define DH_ 64
#define M_ (B_ * S_)      // 16384 sequence rows
#define OUT_ 128

typedef __bf16 bf16_t;
typedef __attribute__((ext_vector_type(16))) __bf16 v16bf;
typedef __attribute__((ext_vector_type(8)))  float  v8f;
typedef __attribute__((ext_vector_type(4)))  unsigned int u32x4;
typedef __attribute__((ext_vector_type(8)))  int  i32x8;
typedef __attribute__((ext_vector_type(4)))  int  i32x4;

// Tensor Data Mover availability (guarded: falls back to manual staging).
#if __has_builtin(__builtin_amdgcn_tensor_load_to_lds)
#define HAVE_TDM 1
#else
#define HAVE_TDM 0
#endif

// ---------------------------------------------------------------------------
// Small helpers
// ---------------------------------------------------------------------------
__device__ inline v8f zero8() {
    v8f z;
#pragma unroll
    for (int i = 0; i < 8; ++i) z[i] = 0.0f;
    return z;
}

// Pack a 16x32 (bf16) operand fragment from an LDS tile laid out
// [rows][32 k] with row stride `stride` (elements).
// CDNA5 16-bit A layout: lane l holds row l%16;
// VGPR v holds K = (v<4?0:16) + (l>=16?8:0) + (v%4)*2 + {0,1}.
__device__ inline v16bf load_frag(const bf16_t* lds, int rowBase, int lane, int stride) {
    union { v16bf v; unsigned int u[8]; } f;
    const int lrow  = rowBase + (lane & 15);
    const int khalf = (lane >> 4) * 8;
    const bf16_t* base = lds + lrow * stride;
#pragma unroll
    for (int vi = 0; vi < 8; ++vi) {
        int k = ((vi & 4) << 2) + khalf + ((vi & 3) << 1);
        f.u[vi] = *(const unsigned int*)(base + k);
    }
    return f.v;
}

// Same fragment, but LDS tile is stored [K rows][N cols] (row stride `stride`),
// i.e. transposed relative to what the operand wants.
__device__ inline v16bf load_frag_t(const bf16_t* lds, int colBase, int lane, int stride) {
    union { v16bf v; unsigned short u[16]; } f;
    const int n     = colBase + (lane & 15);
    const int khalf = (lane >> 4) * 8;
#pragma unroll
    for (int e = 0; e < 16; ++e) {
        int vi = e >> 1, sub = e & 1;
        int k = ((vi & 4) << 2) + khalf + ((vi & 3) << 1) + sub;
        f.u[e] = *(const unsigned short*)(lds + k * stride + n);
    }
    return f.v;
}

#if HAVE_TDM
__device__ inline i32x4 zero_i4() { i32x4 z; for (int i = 0; i < 4; ++i) z[i] = 0; return z; }
__device__ inline i32x8 zero_i8() { i32x8 z; for (int i = 0; i < 8; ++i) z[i] = 0; return z; }

// Issue one TDM tile load: 64 rows x 32 bf16 from a row-major tensor
// (row stride `strideElems`), into LDS at byte offset ldsOff with hardware
// padding of 4 dwords every 16 dwords (=> 40-element LDS row stride).
// D# layout per CDNA5 ISA 8.3/8.4.
__device__ inline void tdm_load_tile(const bf16_t* gsrc, unsigned ldsOff,
                                     int kLen, int rows, int strideElems) {
    unsigned long long ga = (unsigned long long)gsrc;
    u32x4 g0;
    g0[0] = 1u;                                          // count=1, user desc
    g0[1] = ldsOff;                                      // lds_addr (bytes)
    g0[2] = (unsigned)ga;                                // global_addr[31:0]
    g0[3] = ((unsigned)(ga >> 32) & 0x01FFFFFFu)         // global_addr[56:32]
          | 0x80000000u;                                 // type=2 ("image")
    i32x8 g1 = zero_i8();
    g1[0] = (int)((1u << 16)                             // data_size = 2 bytes
                | (1u << 20)                             // pad_enable
                | (3u << 22)                             // pad_interval: 16 dwords
                | (3u << 25));                           // pad_amount: 4 dwords
    g1[1] = (int)(((unsigned)kLen & 0xFFFFu) << 16);     // tensor_dim0[15:0]
    g1[2] = (int)(((unsigned)kLen >> 16)
                | (((unsigned)rows & 0xFFFFu) << 16));   // dim0[31:16] | dim1[15:0]
    g1[3] = (int)((((unsigned)rows >> 16) & 0xFFFFu)
                | (32u << 16));                          // dim1[31:16] | tile_dim0=32
    g1[4] = 64;                                          // tile_dim1 = 64 rows
    g1[5] = strideElems;                                 // tensor_dim0_stride[31:0]
    i32x4 z4 = zero_i4();
#if __has_include(<hip/amd_detail/amd_gfx1250_TDM.h>)
    i32x8 z8 = zero_i8();                                // therock toolchain: 6-arg form
    __builtin_amdgcn_tensor_load_to_lds(g0, g1, z4, z4, z8, 0);
#else
    __builtin_amdgcn_tensor_load_to_lds(g0, g1, z4, z4, 0); // ROCm 7.2: 5-arg form
#endif
}

__device__ inline void wait_tensor0() {
#if __has_builtin(__builtin_amdgcn_s_wait_tensorcnt)
    __builtin_amdgcn_s_wait_tensorcnt(0);
#else
    asm volatile("s_wait_tensorcnt 0x0" ::: "memory");
#endif
}
#endif  // HAVE_TDM

// ---------------------------------------------------------------------------
// Generic tiled WMMA GEMM:  Y[M,N] = act( X[M,K] * B + bias[N] )
//   bmode 0: Bm is W[N,K] row-major (compute X * W^T)
//   bmode 1: Bm is G[K,N] row-major (compute X * G)
// X, Bm bf16; outputs optionally f32 (outF) and/or bf16 (outB).
// Block = 128 threads (4 waves). Tile 64(M) x 64(N), K stepped by 32.
// Requires N % 64 == 0, K % 32 == 0 (true for every GEMM in this model).
// ---------------------------------------------------------------------------
#define LDA_ 40   // padded LDS row stride (elements) for [rows][k] tiles
#define LDT_ 72   // padded LDS row stride for the transposed [k][n] tile

__global__ void gemm_wmma(const bf16_t* __restrict__ X, int ldx,
                          const bf16_t* __restrict__ Bm, int ldb,
                          const float* __restrict__ bias,
                          float* __restrict__ outF, bf16_t* __restrict__ outB, int ldo,
                          int Mdim, int Ndim, int Kdim, int bmode, int act) {
    __shared__ bf16_t As[64 * LDA_];
    __shared__ bf16_t Bs[64 * LDA_];   // also holds 32*LDT_ = 2304 <= 2560 for bmode 1

    const int tid  = threadIdx.x;
    const int lane = tid & 31;
    const int wv   = tid >> 5;          // 0..3 : N sub-tile of this wave
    const int m0   = blockIdx.x * 64;
    const int n0   = blockIdx.y * 64;

    v8f acc[4];
#pragma unroll
    for (int i = 0; i < 4; ++i) acc[i] = zero8();

    const int r    = tid >> 1;          // 0..63 : tile row this thread stages
    const int half = tid & 1;           // 0/1   : low/high 16 of the k-chunk

#if HAVE_TDM
    const unsigned ldsAs = (unsigned)(unsigned long long)(const void*)As;
    const unsigned ldsBs = (unsigned)(unsigned long long)(const void*)Bs;
#endif

    for (int k0 = 0; k0 < Kdim; k0 += 32) {
        __syncthreads();

#if HAVE_TDM
        // ---- stage A tile (and B tile for bmode 0) via the Tensor Data Mover.
        // TDM ignores EXEC, so only wave 0 may reach the instruction.
        if (tid < 32) {
            tdm_load_tile(X + (size_t)m0 * ldx + k0, ldsAs, Kdim, Mdim - m0, ldx);
            if (bmode == 0)
                tdm_load_tile(Bm + (size_t)n0 * ldb + k0, ldsBs, Kdim, Ndim - n0, ldb);
            wait_tensor0();
        }
#else
        // ---- stage A tile (64 x 32), coalesced 16B loads ----
        {
            int gr = m0 + r;
            bf16_t* dst = &As[r * LDA_ + half * 16];
            if (gr < Mdim) {
                const bf16_t* src = X + (size_t)gr * ldx + k0 + half * 16;
                __builtin_prefetch(src + 32, 0, 1);   // global_prefetch_b8
                *(uint4*)dst       = *(const uint4*)src;
                *(uint4*)(dst + 8) = *(const uint4*)(src + 8);
            } else {
#pragma unroll
                for (int e = 0; e < 16; ++e) dst[e] = (bf16_t)0.0f;
            }
        }
        if (bmode == 0) {
            const bf16_t* src = Bm + (size_t)(n0 + r) * ldb + k0 + half * 16;
            __builtin_prefetch(src + 32, 0, 1);
            bf16_t* dst = &Bs[r * LDA_ + half * 16];
            *(uint4*)dst       = *(const uint4*)src;
            *(uint4*)(dst + 8) = *(const uint4*)(src + 8);
        }
#endif
        if (bmode != 0) {
            // ---- stage B tile in native [K=32][N=64] order, coalesced 16B loads
            int kk = tid >> 2, q = tid & 3;
            const bf16_t* src = Bm + (size_t)(k0 + kk) * ldb + n0 + q * 16;
            __builtin_prefetch(src + 32 * ldb, 0, 1);
            *(uint4*)&Bs[kk * LDT_ + q * 16] = *(const uint4*)src;
        }
        __syncthreads();

        // ---- compute: each wave does 4 WMMAs (4 M sub-tiles x its N sub-tile)
        v16bf bfrag = (bmode == 0) ? load_frag(Bs, wv * 16, lane, LDA_)
                                   : load_frag_t(Bs, wv * 16, lane, LDT_);
#pragma unroll
        for (int mt = 0; mt < 4; ++mt) {
            v16bf afrag = load_frag(As, mt * 16, lane, LDA_);
            acc[mt] = __builtin_amdgcn_wmma_f32_16x16x32_bf16(
                false, afrag, false, bfrag, (short)0, acc[mt], false, false);
        }
    }

    // ---- epilogue: C/D layout -> lane holds col = lane%16, rows r + (lane>=16?8:0)
    const int col  = n0 + wv * 16 + (lane & 15);
    const int rsel = (lane >> 4) * 8;
    const float bv = bias ? bias[col] : 0.0f;
#pragma unroll
    for (int mt = 0; mt < 4; ++mt) {
#pragma unroll
        for (int rr = 0; rr < 8; ++rr) {
            int row = m0 + mt * 16 + rsel + rr;
            if (row < Mdim) {
                float val = acc[mt][rr] + bv;
                if (act) val = fmaxf(val, 0.0f);
                if (outF) outF[(size_t)row * ldo + col] = val;
                if (outB) outB[(size_t)row * ldo + col] = (bf16_t)val;
            }
        }
    }
}

// ---------------------------------------------------------------------------
// GRU gate nonlinearity (one step).
// ---------------------------------------------------------------------------
__global__ void gru_gates(const float* __restrict__ xi, int t,
                          const float* __restrict__ gh,
                          float* __restrict__ h, bf16_t* __restrict__ hb,
                          bf16_t* __restrict__ seqB) {
    int i = blockIdx.x * blockDim.x + threadIdx.x;
    if (i >= B_ * H_) return;
    int b = i >> 9, j = i & (H_ - 1);
    size_t xrow = ((size_t)b * S_ + t) * H3_;
    float xr = xi[xrow + j], xz = xi[xrow + H_ + j], xn = xi[xrow + 2 * H_ + j];
    size_t grow = (size_t)b * H3_;
    float hr = gh[grow + j], hz = gh[grow + H_ + j], hn = gh[grow + 2 * H_ + j];
    float rg = 1.0f / (1.0f + expf(-(xr + hr)));
    float zg = 1.0f / (1.0f + expf(-(xz + hz)));
    float ng = tanhf(xn + rg * hn);
    float hnew = (1.0f - zg) * ng + zg * h[i];
    h[i]  = hnew;
    hb[i] = (bf16_t)hnew;
    if (seqB) seqB[((size_t)b * S_ + t) * H_ + j] = (bf16_t)hnew;
}

// ---------------------------------------------------------------------------
// LayerNorm(a + res) over H=512, one wave per row (wave32).
// ---------------------------------------------------------------------------
__global__ void ln_residual(const float* __restrict__ a, const float* __restrict__ res,
                            const float* __restrict__ g, const float* __restrict__ beta,
                            float* __restrict__ outF, bf16_t* __restrict__ outB, int rows) {
    int wave = threadIdx.x >> 5, lane = threadIdx.x & 31;
    int row = blockIdx.x * 8 + wave;
    if (row >= rows) return;
    const float* pa = a   + (size_t)row * H_;
    const float* pr = res + (size_t)row * H_;
    float v[16], s = 0.0f;
#pragma unroll
    for (int i = 0; i < 16; ++i) { v[i] = pa[lane + 32 * i] + pr[lane + 32 * i]; s += v[i]; }
#pragma unroll
    for (int off = 16; off; off >>= 1) s += __shfl_xor(s, off, 32);
    float mean = s * (1.0f / H_);
    float q = 0.0f;
#pragma unroll
    for (int i = 0; i < 16; ++i) { float d = v[i] - mean; q += d * d; }
#pragma unroll
    for (int off = 16; off; off >>= 1) q += __shfl_xor(q, off, 32);
    float inv = rsqrtf(q * (1.0f / H_) + 1e-5f);
#pragma unroll
    for (int i = 0; i < 16; ++i) {
        int c = lane + 32 * i;
        float o = (v[i] - mean) * inv * g[c] + beta[c];
        if (outF) outF[(size_t)row * H_ + c] = o;
        if (outB) outB[(size_t)row * H_ + c] = (bf16_t)o;
    }
}

// ---------------------------------------------------------------------------
// Row softmax of (scores / 8) over 512 cols, one wave per row, bf16 out.
// ---------------------------------------------------------------------------
__global__ void softmax_rows(const float* __restrict__ sc, bf16_t* __restrict__ scb) {
    int wave = threadIdx.x >> 5, lane = threadIdx.x & 31;
    int row = blockIdx.x * 8 + wave;               // 512 rows total
    const float* p = sc + (size_t)row * S_;
    float v[16], m = -1e30f;
#pragma unroll
    for (int i = 0; i < 16; ++i) { v[i] = p[lane + 32 * i]; m = fmaxf(m, v[i]); }
#pragma unroll
    for (int off = 16; off; off >>= 1) m = fmaxf(m, __shfl_xor(m, off, 32));
    float s = 0.0f;
#pragma unroll
    for (int i = 0; i < 16; ++i) { v[i] = expf((v[i] - m) * 0.125f); s += v[i]; }
#pragma unroll
    for (int off = 16; off; off >>= 1) s += __shfl_xor(s, off, 32);
    float inv = 1.0f / s;
#pragma unroll
    for (int i = 0; i < 16; ++i)
        scb[(size_t)row * S_ + lane + 32 * i] = (bf16_t)(v[i] * inv);
}

// ---------------------------------------------------------------------------
// Elementwise utilities
// ---------------------------------------------------------------------------
__global__ void cvt_f32_bf16(const float* __restrict__ s, bf16_t* __restrict__ d, int n) {
    int i = blockIdx.x * blockDim.x + threadIdx.x;
    if (i < n) d[i] = (bf16_t)s[i];
}
__global__ void zero_f32(float* p, int n) {
    int i = blockIdx.x * blockDim.x + threadIdx.x;
    if (i < n) p[i] = 0.0f;
}
__global__ void zero_bf16(bf16_t* p, int n) {
    int i = blockIdx.x * blockDim.x + threadIdx.x;
    if (i < n) p[i] = (bf16_t)0.0f;
}
__global__ void copy_h_to_cat(const float* __restrict__ h, bf16_t* __restrict__ cat, int off) {
    int i = blockIdx.x * blockDim.x + threadIdx.x;
    if (i >= B_ * H_) return;
    int b = i >> 9, j = i & (H_ - 1);
    cat[(size_t)b * (2 * H_) + off + j] = (bf16_t)h[i];
}

// ---------------------------------------------------------------------------
// Host orchestration
// ---------------------------------------------------------------------------
static inline size_t align256(size_t x) { return (x + 255) & ~(size_t)255; }

enum {
    IN_FC = 0, IN_MG = 1,
    ENC_FC = 2, ENC_MG = 6,
    ATTN1 = 10, ATTN2 = 14,
    GRU1 = 18, GRU2 = 22, GRU3 = 26, GRU4 = 30,
    LIN1 = 34, LIN2 = 40,
    FCM = 46,
    LN1G = 52, LN1B = 53, LN2G = 54, LN2B = 55
};

extern "C" void kernel_launch(void* const* d_in, const int* in_sizes, int n_in,
                              void* d_out, int out_size, void* d_ws, size_t ws_size,
                              hipStream_t stream) {
    char* ws = (char*)d_ws;
    size_t off = 0;
    auto alloc = [&](size_t bytes) { size_t o = off; off = align256(off + bytes); return o; };

    struct WEnt { int idx; size_t n; };
    static const WEnt wl[] = {
        {ENC_FC, 786432}, {ENC_FC + 1, 786432}, {ENC_MG, 786432}, {ENC_MG + 1, 786432},
        {ATTN1, 786432}, {ATTN1 + 2, 262144}, {ATTN2, 786432}, {ATTN2 + 2, 262144},
        {GRU1, 786432}, {GRU1 + 1, 786432}, {GRU2, 786432}, {GRU2 + 1, 786432},
        {GRU3, 786432}, {GRU3 + 1, 786432}, {GRU4, 786432}, {GRU4 + 1, 786432},
        {LIN1, 524288}, {LIN1 + 2, 1048576}, {LIN1 + 4, 524288},
        {LIN2, 524288}, {LIN2 + 2, 1048576}, {LIN2 + 4, 524288},
        {FCM, 1048576}, {FCM + 2, 1048576}, {FCM + 4, 131072}
    };
    size_t woff[56] = {0};
    for (auto& w : wl) {
        woff[w.idx] = alloc(w.n * sizeof(bf16_t));
        cvt_f32_bf16<<<(int)((w.n + 255) / 256), 256, 0, stream>>>(
            (const float*)d_in[w.idx], (bf16_t*)(ws + woff[w.idx]), (int)w.n);
    }
    auto WB = [&](int idx) { return (const bf16_t*)(ws + woff[idx]); };
    auto F  = [&](int idx) { return (const float*)d_in[idx]; };

    size_t oXB  = alloc((size_t)M_ * H_ * 2);
    size_t oXI  = alloc((size_t)M_ * H3_ * 4);
    size_t oS1  = alloc((size_t)M_ * H_ * 4);
    size_t oS1B = alloc((size_t)M_ * H_ * 2);
    size_t oQKV = alloc((size_t)M_ * H3_ * 2);
    size_t oA   = alloc((size_t)M_ * H_ * 4);
    size_t oAB  = alloc((size_t)M_ * H_ * 2);
    size_t oS2B = alloc((size_t)M_ * H_ * 2);
    size_t oSC  = alloc((size_t)S_ * S_ * 4);
    size_t oSCB = alloc((size_t)S_ * S_ * 2);
    size_t oH   = alloc((size_t)B_ * H_ * 4);
    size_t oHB  = alloc((size_t)B_ * H_ * 2);
    size_t oGH  = alloc((size_t)B_ * H3_ * 4);
    size_t oCAT = alloc((size_t)B_ * 2 * H_ * 2);
    size_t oT1  = alloc((size_t)B_ * 1024 * 2);
    size_t oT2  = alloc((size_t)B_ * 1024 * 2);
    (void)ws_size; (void)in_sizes; (void)n_in; (void)out_size;

    float*  XI  = (float*)(ws + oXI);
    bf16_t* M1B = (bf16_t*)(ws + oXI);
    bf16_t* M2B = (bf16_t*)(ws + oXI + align256((size_t)M_ * 1024 * 2));
    float*  S1  = (float*)(ws + oS1);
    bf16_t* S1B = (bf16_t*)(ws + oS1B);
    bf16_t* XB  = (bf16_t*)(ws + oXB);
    bf16_t* QKV = (bf16_t*)(ws + oQKV);
    float*  A   = (float*)(ws + oA);
    bf16_t* AB  = (bf16_t*)(ws + oAB);
    bf16_t* S2B = (bf16_t*)(ws + oS2B);
    float*  SC  = (float*)(ws + oSC);
    bf16_t* SCB = (bf16_t*)(ws + oSCB);
    float*  Hh  = (float*)(ws + oH);
    bf16_t* HB  = (bf16_t*)(ws + oHB);
    float*  GH  = (float*)(ws + oGH);
    bf16_t* CAT = (bf16_t*)(ws + oCAT);
    bf16_t* T1B = (bf16_t*)(ws + oT1);
    bf16_t* T2B = (bf16_t*)(ws + oT2);

    auto gemm = [&](const bf16_t* X, int ldx, const bf16_t* Bm, int ldb,
                    const float* bias, float* oF, bf16_t* oB, int ldo,
                    int Md, int Nd, int Kd, int bmode, int act) {
        dim3 g((Md + 63) / 64, Nd / 64);
        gemm_wmma<<<g, 128, 0, stream>>>(X, ldx, Bm, ldb, bias, oF, oB, ldo,
                                         Md, Nd, Kd, bmode, act);
    };
    auto scan = [&](int gbase, bf16_t* seqB) {
        for (int t = 0; t < S_; ++t) {
            gemm(HB, H_, WB(gbase + 1), H_, F(gbase + 3), GH, nullptr, H3_,
                 B_, H3_, H_, 0, 0);
            gru_gates<<<(B_ * H_) / 256, 256, 0, stream>>>(XI, t, GH, Hh, HB, seqB);
        }
    };

    auto branch = [&](int xIdx, int encB, int attnB, int g1B, int g2B, int linB,
                      int lnG, int lnBi, int catOff) {
        const float* x = F(xIdx);
        cvt_f32_bf16<<<(M_ * H_) / 256, 256, 0, stream>>>(x, XB, M_ * H_);
        gemm(XB, H_, WB(encB), H_, F(encB + 2), XI, nullptr, H3_, M_, H3_, H_, 0, 0);
        zero_f32<<<(B_ * H_) / 256, 256, 0, stream>>>(Hh, B_ * H_);
        zero_bf16<<<(B_ * H_) / 256, 256, 0, stream>>>(HB, B_ * H_);
        scan(encB, S1B);
        gemm(S1B, H_, WB(attnB), H_, F(attnB + 1), nullptr, QKV, H3_, M_, H3_, H_, 0, 0);
        for (int b = 0; b < B_; ++b) {
            for (int hd = 0; hd < NH_; ++hd) {
                const bf16_t* q = QKV + (size_t)b * S_ * H3_ + hd * DH_;
                const bf16_t* k = q + H_;
                const bf16_t* v = q + 2 * H_;
                gemm(q, H3_, k, H3_, nullptr, SC, nullptr, S_, S_, S_, DH_, 0, 0);
                softmax_rows<<<S_ / 8, 256, 0, stream>>>(SC, SCB);
                gemm(SCB, S_, v, H3_, nullptr, nullptr,
                     S1B + (size_t)b * S_ * H_ + hd * DH_, H_, S_, DH_, S_, 1, 0);
            }
        }
        gemm(S1B, H_, WB(attnB + 2), H_, F(attnB + 3), S1, nullptr, H_, M_, H_, H_, 0, 0);
        ln_residual<<<M_ / 8, 256, 0, stream>>>(S1, x, F(lnG), F(lnBi), A, AB, M_);
        gemm(AB, H_, WB(g1B), H_, F(g1B + 2), XI, nullptr, H3_, M_, H3_, H_, 0, 0);
        scan(g1B, S2B);
        gemm(S2B, H_, WB(linB), H_, F(linB + 1), nullptr, M1B, 1024, M_, 1024, H_, 0, 1);
        gemm(M1B, 1024, WB(linB + 2), 1024, F(linB + 3), nullptr, M2B, 1024, M_, 1024, 1024, 0, 1);
        gemm(M2B, 1024, WB(linB + 4), 1024, F(linB + 5), S1, nullptr, H_, M_, H_, 1024, 0, 0);
        ln_residual<<<M_ / 8, 256, 0, stream>>>(S1, A, F(lnG), F(lnBi), nullptr, S1B, M_);
        gemm(S1B, H_, WB(g2B), H_, F(g2B + 2), XI, nullptr, H3_, M_, H3_, H_, 0, 0);
        scan(g2B, nullptr);
        copy_h_to_cat<<<(B_ * H_) / 256, 256, 0, stream>>>(Hh, CAT, catOff);
    };

    branch(IN_FC, ENC_FC, ATTN1, GRU1, GRU2, LIN1, LN1G, LN1B, 0);
    branch(IN_MG, ENC_MG, ATTN2, GRU3, GRU4, LIN2, LN2G, LN2B, H_);

    gemm(CAT, 1024, WB(FCM), 1024, F(FCM + 1), nullptr, T1B, 1024, B_, 1024, 1024, 0, 1);
    gemm(T1B, 1024, WB(FCM + 2), 1024, F(FCM + 3), nullptr, T2B, 1024, B_, 1024, 1024, 0, 1);
    gemm(T2B, 1024, WB(FCM + 4), 1024, F(FCM + 5), (float*)d_out, nullptr, OUT_,
         B_, OUT_, 1024, 0, 0);
}
